// HGEncoder_76373108457758
// MI455X (gfx1250) — matrix operand order repaired and checked by reading, not compiled, not run on hardware
//
#include <hip/hip_runtime.h>
#include <hip/hip_bf16.h>
#include <cmath>

// ---------------------------------------------------------------------------
// HGEncoder for MI455X (gfx1250, wave32).
// SpMM layers are HBM/atomic bound (~9 GB traffic -> global_load_b128 +
// global_atomic_add_f32 + global_prefetch). Dense projections and the final
// ue_a @ all_a^T run on v_wmma_f32_16x16x32_bf16 (f32 accumulate), with the
// shared B tile staged into LDS by the Tensor Data Mover (tensor_load_to_lds).
// ---------------------------------------------------------------------------

#define NU 50000
#define NL 25000
#define NT 2048
#define NA 4096
#define DD 64
#define NLAYERS 3
#define NB 4096
#define NNZ_U 800000
#define NNZ_L 400000
#define NNZ_S 65536

typedef __attribute__((ext_vector_type(16))) __bf16 v16bf;
typedef __attribute__((ext_vector_type(8)))  __bf16 v8bf;
typedef __attribute__((ext_vector_type(8)))  float  v8f;
typedef __attribute__((ext_vector_type(4)))  unsigned int v4u;
typedef __attribute__((ext_vector_type(8)))  unsigned int v8u;

// ---------------- elementwise helpers ----------------
__global__ void k_zero(float* __restrict__ p, int n) {
  int i = blockIdx.x * blockDim.x + threadIdx.x;
  if (i < n) p[i] = 0.0f;
}
__global__ void k_copy(float* __restrict__ d, const float* __restrict__ s, int n) {
  int i = blockIdx.x * blockDim.x + threadIdx.x;
  if (i < n) d[i] = s[i];
}
__global__ void k_add(float* __restrict__ d, const float* __restrict__ s, int n) {
  int i = blockIdx.x * blockDim.x + threadIdx.x;
  if (i < n) d[i] += s[i];
}
__global__ void k_scale(float* __restrict__ p, float a, int n) {
  int i = blockIdx.x * blockDim.x + threadIdx.x;
  if (i < n) p[i] *= a;
}
__global__ void k_cvt_bf16(const float* __restrict__ s, __bf16* __restrict__ d, int n) {
  int i = blockIdx.x * blockDim.x + threadIdx.x;
  if (i < n) d[i] = (__bf16)s[i];
}

// ---------------- COO SpMM: Y[row] += val * X[col], D=64 ----------------
// 16 lanes per nonzero, each lane moves one float4 (whole row = 256B coalesced).
// Prefetch the gathered row ~512 entries ahead to hide random-gather latency.
__global__ void k_spmm_atomic(const int* __restrict__ row, const int* __restrict__ col,
                              const float* __restrict__ val, const float* __restrict__ X,
                              float* __restrict__ Y, int nnz) {
  int t = blockIdx.x * blockDim.x + threadIdx.x;
  int e = t >> 4;
  int part = t & 15;
  if (e >= nnz) return;
  if (e + 512 < nnz) {
    int c2 = col[e + 512];
    __builtin_prefetch(X + (size_t)c2 * DD + part * 4, 0, 0);
  }
  int r = row[e];
  int c = col[e];
  float v = val[e];
  float4 x = ((const float4*)(X + (size_t)c * DD))[part];
  float* y = Y + (size_t)r * DD + part * 4;
  unsafeAtomicAdd(y + 0, v * x.x);
  unsafeAtomicAdd(y + 1, v * x.y);
  unsafeAtomicAdd(y + 2, v * x.z);
  unsafeAtomicAdd(y + 3, v * x.w);
}

// ---------------- Out[M x 64] = X[M x 64] @ W^T + bias (WMMA bf16) ----------
// One wave32 per 16x16 tile; K=64 as two 16x16x32 bf16 WMMA steps.
__global__ void k_proj_wmma(const float* __restrict__ X, const float* __restrict__ W,
                            const float* __restrict__ bias, float* __restrict__ Out, int M) {
  int wave = (blockIdx.x * blockDim.x + threadIdx.x) >> 5;
  int lane = threadIdx.x & 31;
  int tile_m = wave >> 2;          // 64/16 = 4 tiles along N
  int tile_n = wave & 3;
  if (tile_m * 16 >= M) return;    // wave-uniform: EXEC stays all-ones for WMMA
  int rowA = tile_m * 16 + (lane & 15);
  int colB = tile_n * 16 + (lane & 15);
  int kbA = (lane >> 4) * 8;
  int kbB = (lane >> 4) * 8;
  v8f c = {};
#pragma unroll
  for (int s = 0; s < 2; ++s) {
    int k0 = s * 32;
    v16bf a, b;
    const float* ap = X + (size_t)rowA * DD + k0 + kbA;
#pragma unroll
    for (int i = 0; i < 8; ++i) a[i] = (__bf16)ap[i];
#pragma unroll
    for (int i = 0; i < 8; ++i) a[8 + i] = (__bf16)ap[16 + i];
    const float* bp = W + (size_t)colB * DD + k0 + kbB;
#pragma unroll
    for (int i = 0; i < 8; ++i) b[i] = (__bf16)bp[i];
#pragma unroll
    for (int i = 0; i < 8; ++i) b[8 + i] = (__bf16)bp[16 + i];
    c = __builtin_amdgcn_wmma_f32_16x16x32_bf16(false, a, false, b, (short)0, c,
                                                false, false);
  }
  int n = tile_n * 16 + (lane & 15);
  float bv = bias[n];
  int mbase = tile_m * 16 + 8 * (lane >> 4);
#pragma unroll
  for (int r = 0; r < 8; ++r)
    Out[(size_t)(mbase + r) * DD + n] = c[r] + bv;
}

// ---------------- per-batch scalar + ue_a gather/pack -----------------------
__global__ void k_score_prep(const int* __restrict__ users, const int* __restrict__ locs,
                             const int* __restrict__ tims,
                             const float* __restrict__ uL, const float* __restrict__ eL,
                             const float* __restrict__ uT, const float* __restrict__ eT,
                             const float* __restrict__ uA,
                             __bf16* __restrict__ ueAb, float* __restrict__ srow) {
  int wave = (blockIdx.x * blockDim.x + threadIdx.x) >> 5;
  int lane = threadIdx.x & 31;
  if (wave >= NB) return;
  int u = users[wave], l = locs[wave], t = tims[wave];
  float2 x1 = ((const float2*)(uL + (size_t)u * DD))[lane];
  float2 y1 = ((const float2*)(eL + (size_t)l * DD))[lane];
  float2 x2 = ((const float2*)(uT + (size_t)u * DD))[lane];
  float2 y2 = ((const float2*)(eT + (size_t)t * DD))[lane];
  float s = x1.x * y1.x + x1.y * y1.y + x2.x * y2.x + x2.y * y2.y;
#pragma unroll
  for (int off = 16; off > 0; off >>= 1) s += __shfl_xor(s, off, 32);
  if (lane == 0) srow[wave] = s;
  float2 x3 = ((const float2*)(uA + (size_t)u * DD))[lane];
  ueAb[(size_t)wave * DD + 2 * lane]     = (__bf16)x3.x;
  ueAb[(size_t)wave * DD + 2 * lane + 1] = (__bf16)x3.y;
}

// ---------------- Out[b,j] = sigmoid(srow[b] + ue_a[b] . all_a[j]) ----------
// Block = 8 waves sharing one tile_n. The 16x64 bf16 B tile (2048B contiguous)
// is staged into LDS once per workgroup by the Tensor Data Mover; waves read
// B fragments from LDS (ds_load_b128) and A fragments from global.
__global__ void k_score_gemm(const __bf16* __restrict__ Abf,   // [NB x 64]
                             const __bf16* __restrict__ Bbf,   // [NA x 64]
                             const float* __restrict__ srow,
                             float* __restrict__ Out) {        // [NB x NA]
  __shared__ __align__(16) __bf16 Bsh[16 * DD];
  int wave = threadIdx.x >> 5;
  int lane = threadIdx.x & 31;
  int tile_n = blockIdx.x & (NA / 16 - 1);   // 0..255
  int mgroup = blockIdx.x >> 8;              // 0..31
  int tile_m = mgroup * 8 + wave;

  if (wave == 0) {
    // Build a 1-D TDM descriptor (D#): 512 dwords (2048B) Global -> LDS.
    unsigned long long ga =
        (unsigned long long)(const void*)(Bbf + (size_t)tile_n * 16 * DD);
    unsigned lds = (unsigned)(uintptr_t)(void*)Bsh;  // low 32 bits = LDS offset
    v4u g0;
    g0[0] = 1u;                                            // count=1, flags=0
    g0[1] = lds;                                           // lds_addr
    g0[2] = (unsigned)ga;                                  // global_addr[31:0]
    g0[3] = (unsigned)((ga >> 32) & 0x01ffffffu) | (2u << 30); // addr hi, type=2
    v8u g1;
    g1[0] = (2u << 16);      // workgroup_mask=0, data_size=4B
    g1[1] = (512u << 16);    // tensor_dim0 = 512 dwords (lo16)
    g1[2] = (1u << 16);      // tensor_dim0 hi = 0, tensor_dim1 = 1 (lo16)
    g1[3] = (512u << 16);    // tensor_dim1 hi = 0, tile_dim0 = 512
    g1[4] = 1u;              // tile_dim1 = 1, tile_dim2 = 0
    g1[5] = 512u;            // tensor_dim0_stride (lo32) = 512
    g1[6] = 0u;              // stride0 hi, stride1 lo
    g1[7] = 0u;              // stride1 hi
    asm volatile("tensor_load_to_lds %0, %1" : : "s"(g0), "s"(g1) : "memory");
    __builtin_amdgcn_s_wait_tensorcnt(0);
  }
  __syncthreads();

  int rowA = tile_m * 16 + (lane & 15);
  int kbA = (lane >> 4) * 8;
  int colB = lane & 15;
  int kbB = (lane >> 4) * 8;
  v8f c = {};
#pragma unroll
  for (int s = 0; s < 2; ++s) {
    int k0 = s * 32;
    const __bf16* ap = Abf + (size_t)rowA * DD + k0 + kbA;
    v8bf alo = *(const v8bf*)(ap);
    v8bf ahi = *(const v8bf*)(ap + 16);
    v16bf a = __builtin_shufflevector(alo, ahi, 0, 1, 2, 3, 4, 5, 6, 7,
                                      8, 9, 10, 11, 12, 13, 14, 15);
    const __bf16* bp = Bsh + colB * DD + k0 + kbB;   // LDS (ds_load_b128)
    v8bf blo = *(const v8bf*)(bp);
    v8bf bhi = *(const v8bf*)(bp + 16);
    v16bf b = __builtin_shufflevector(blo, bhi, 0, 1, 2, 3, 4, 5, 6, 7,
                                      8, 9, 10, 11, 12, 13, 14, 15);
    c = __builtin_amdgcn_wmma_f32_16x16x32_bf16(false, a, false, b, (short)0, c,
                                                false, false);
  }
  int n = tile_n * 16 + (lane & 15);
  int mbase = tile_m * 16 + 8 * (lane >> 4);
#pragma unroll
  for (int r = 0; r < 8; ++r) {
    float sc = c[r] + srow[mbase + r];
    Out[(size_t)(mbase + r) * NA + n] = 1.0f / (1.0f + __expf(-sc));
  }
}

// ---------------------------------------------------------------------------
extern "C" void kernel_launch(void* const* d_in, const int* in_sizes, int n_in,
                              void* d_out, int out_size, void* d_ws, size_t ws_size,
                              hipStream_t stream) {
  (void)in_sizes; (void)n_in; (void)out_size; (void)ws_size;
  auto F = [&](int i) { return (const float*)d_in[i]; };
  auto I = [&](int i) { return (const int*)d_in[i]; };

  // sparse triples (row, col, val) in setup_inputs order
  const int *HGul_r = I(0),  *HGul_c = I(1);  const float *HGul_v = F(2);
  const int *HGut_r = I(3),  *HGut_c = I(4);  const float *HGut_v = F(5);
  const int *HGua_r = I(6),  *HGua_c = I(7);  const float *HGua_v = F(8);
  const int *HGl_r  = I(9),  *HGl_c  = I(10); const float *HGl_v  = F(11);
  const int *HGt_r  = I(12), *HGt_c  = I(13); const float *HGt_v  = F(14);
  const int *HGa_r  = I(15), *HGa_c  = I(16); const float *HGa_v  = F(17);
  const int *Vlu_r  = I(18), *Vlu_c  = I(19); const float *Vlu_v  = F(20);
  const int *Vtu_r  = I(21), *Vtu_c  = I(22); const float *Vtu_v  = F(23);
  const int *Vau_r  = I(24), *Vau_c  = I(25); const float *Vau_v  = F(26);
  const int *Vul_r  = I(27), *Vul_c  = I(28); const float *Vul_v  = F(29);
  const int *Vut_r  = I(30), *Vut_c  = I(31); const float *Vut_v  = F(32);
  const int *Vua_r  = I(33), *Vua_c  = I(34); const float *Vua_v  = F(35);

  const float* user_emb = F(36);
  const float* loc_emb  = F(37);
  const float* time_emb = F(38);
  const float* act_emb  = F(39);
  const float *wl = F(40), *bl = F(41);
  const float *wt = F(42), *bt = F(43);
  const float *wa = F(44), *ba = F(45);
  const int *users = I(46), *locs = I(47), *tims = I(48);

  // workspace layout (~141 MB of f32 + 1 MB bf16)
  float* ws = (float*)d_ws;
  size_t off = 0;
  auto take = [&](size_t n) { float* p = ws + off; off += n; return p; };
  const size_t U = (size_t)NU * DD, L = (size_t)NL * DD;
  const size_t T = (size_t)NT * DD, A = (size_t)NA * DD;
  float *uLc = take(U), *uTc = take(U), *uAc = take(U);
  float *uLn = take(U), *uTn = take(U), *uAn = take(U);
  float *uLa = take(U), *uTa = take(U), *uAa = take(U);
  float *eLc = take(L), *eLn = take(L), *eLa = take(L);
  float *eTc = take(T), *eTn = take(T), *eTa = take(T);
  float *eAc = take(A), *eAn = take(A), *eAa = take(A);
  float *srow = take((size_t)NB);
  __bf16* ueAb  = (__bf16*)(ws + off); off += (size_t)NB * DD / 2;
  __bf16* allAb = (__bf16*)(ws + off); off += A / 2;

  const int TPB = 256;
  auto gs = [](size_t n) { return (unsigned)((n + 255) / 256); };

  // 1) disentangled projections on WMMA
  {
    unsigned tiles = (NU / 16) * 4;
    unsigned blocks = (tiles + 7) / 8;   // 8 waves per 256-thread block
    k_proj_wmma<<<blocks, TPB, 0, stream>>>(user_emb, wl, bl, uLc, NU);
    k_proj_wmma<<<blocks, TPB, 0, stream>>>(user_emb, wt, bt, uTc, NU);
    k_proj_wmma<<<blocks, TPB, 0, stream>>>(user_emb, wa, ba, uAc, NU);
  }
  k_copy<<<gs(L), TPB, 0, stream>>>(eLc, loc_emb, (int)L);
  k_copy<<<gs(T), TPB, 0, stream>>>(eTc, time_emb, (int)T);
  k_copy<<<gs(A), TPB, 0, stream>>>(eAc, act_emb, (int)A);
  // acc = layer-0 state
  k_copy<<<gs(U), TPB, 0, stream>>>(uLa, uLc, (int)U);
  k_copy<<<gs(U), TPB, 0, stream>>>(uTa, uTc, (int)U);
  k_copy<<<gs(U), TPB, 0, stream>>>(uAa, uAc, (int)U);
  k_copy<<<gs(L), TPB, 0, stream>>>(eLa, eLc, (int)L);
  k_copy<<<gs(T), TPB, 0, stream>>>(eTa, eTc, (int)T);
  k_copy<<<gs(A), TPB, 0, stream>>>(eAa, eAc, (int)A);

  // 2) hypergraph propagation layers (memory/atomic bound)
  for (int layer = 0; layer < NLAYERS; ++layer) {
    k_zero<<<gs(U), TPB, 0, stream>>>(uLn, (int)U);
    k_zero<<<gs(U), TPB, 0, stream>>>(uTn, (int)U);
    k_zero<<<gs(U), TPB, 0, stream>>>(uAn, (int)U);
    k_zero<<<gs(L), TPB, 0, stream>>>(eLn, (int)L);
    k_zero<<<gs(T), TPB, 0, stream>>>(eTn, (int)T);
    k_zero<<<gs(A), TPB, 0, stream>>>(eAn, (int)A);

    auto spmm = [&](const int* r, const int* c, const float* v,
                    const float* X, float* Y, int nnz) {
      k_spmm_atomic<<<gs((size_t)nnz * 16), TPB, 0, stream>>>(r, c, v, X, Y, nnz);
    };
    // location channel
    spmm(HGul_r, HGul_c, HGul_v, uLc, uLn, NNZ_U);
    spmm(Vlu_r,  Vlu_c,  Vlu_v,  eLc, uLn, NNZ_U);
    spmm(HGl_r,  HGl_c,  HGl_v,  eLc, eLn, NNZ_L);
    spmm(Vul_r,  Vul_c,  Vul_v,  uLc, eLn, NNZ_L);
    // time channel
    spmm(HGut_r, HGut_c, HGut_v, uTc, uTn, NNZ_U);
    spmm(Vtu_r,  Vtu_c,  Vtu_v,  eTc, uTn, NNZ_U);
    spmm(HGt_r,  HGt_c,  HGt_v,  eTc, eTn, NNZ_S);
    spmm(Vut_r,  Vut_c,  Vut_v,  uTc, eTn, NNZ_S);
    // activity channel
    spmm(HGua_r, HGua_c, HGua_v, uAc, uAn, NNZ_U);
    spmm(Vau_r,  Vau_c,  Vau_v,  eAc, uAn, NNZ_U);
    spmm(HGa_r,  HGa_c,  HGa_v,  eAc, eAn, NNZ_S);
    spmm(Vua_r,  Vua_c,  Vua_v,  uAc, eAn, NNZ_S);

    k_add<<<gs(U), TPB, 0, stream>>>(uLa, uLn, (int)U);
    k_add<<<gs(U), TPB, 0, stream>>>(uTa, uTn, (int)U);
    k_add<<<gs(U), TPB, 0, stream>>>(uAa, uAn, (int)U);
    k_add<<<gs(L), TPB, 0, stream>>>(eLa, eLn, (int)L);
    k_add<<<gs(T), TPB, 0, stream>>>(eTa, eTn, (int)T);
    k_add<<<gs(A), TPB, 0, stream>>>(eAa, eAn, (int)A);

    float* t0;
    t0 = uLc; uLc = uLn; uLn = t0;
    t0 = uTc; uTc = uTn; uTn = t0;
    t0 = uAc; uAc = uAn; uAn = t0;
    t0 = eLc; eLc = eLn; eLn = t0;
    t0 = eTc; eTc = eTn; eTn = t0;
    t0 = eAc; eAc = eAn; eAn = t0;
  }

  // 3) layer mean (sum of LAYERS+1 states -> * 0.25)
  const float inv = 1.0f / (NLAYERS + 1);
  k_scale<<<gs(U), TPB, 0, stream>>>(uLa, inv, (int)U);
  k_scale<<<gs(U), TPB, 0, stream>>>(uTa, inv, (int)U);
  k_scale<<<gs(U), TPB, 0, stream>>>(uAa, inv, (int)U);
  k_scale<<<gs(L), TPB, 0, stream>>>(eLa, inv, (int)L);
  k_scale<<<gs(T), TPB, 0, stream>>>(eTa, inv, (int)T);
  k_scale<<<gs(A), TPB, 0, stream>>>(eAa, inv, (int)A);

  // 4) rating: per-row scalar + bf16 pack, then WMMA GEMM + sigmoid
  k_score_prep<<<NB / 8, TPB, 0, stream>>>(users, locs, tims, uLa, eLa, uTa, eTa,
                                           uAa, ueAb, srow);
  k_cvt_bf16<<<gs(A), TPB, 0, stream>>>(eAa, allAb, (int)A);
  {
    // 8 waves per block share one tile_n; 32 m-groups x 256 n-tiles = 8192 WGs
    unsigned blocks = (NB / 16 / 8) * (NA / 16);
    k_score_gemm<<<blocks, TPB, 0, stream>>>(ueAb, allAb, srow, (float*)d_out);
  }
}
